// HAN_81801947119983
// MI455X (gfx1250) — matrix operand order repaired and checked by reading, not compiled, not run on hardware
//
#include <hip/hip_runtime.h>
#include <hip/hip_bf16.h>
#include <math.h>

// ---------------------------------------------------------------------------
// CDNA5 (gfx1250) HAN implementation, wave32 + v_wmma_f32_16x16x32_f16.
// ---------------------------------------------------------------------------

typedef __attribute__((ext_vector_type(16))) _Float16 v16h;
typedef __attribute__((ext_vector_type(8)))  _Float16 v8h;
typedef __attribute__((ext_vector_type(8)))  float    v8f;

union V16U { v16h v; v8h h[2]; };

// A-matrix 16x32 f16 fragment (ISA 7.12.2): lane row = lane&15,
// lo lanes K = {0..7,16..23}, hi lanes K = {8..15,24..31}.
__device__ __forceinline__ v16h fragA(const _Float16* p, int kgrp) {
  V16U u;
  u.h[0] = *(const v8h*)(p + kgrp * 8);
  u.h[1] = *(const v8h*)(p + 16 + kgrp * 8);
  return u.v;
}
// B-matrix 32x16 f16 fragment: lane col = lane&15, lo lanes K={0..7,16..23}, hi K={8..15,24..31}.
// p points at WT[n] + kc*32 (weights stored transposed: WT[n][k]).
__device__ __forceinline__ v16h fragB(const _Float16* p, int khalf) {
  V16U u;
  u.h[0] = *(const v8h*)(p + khalf * 8);
  u.h[1] = *(const v8h*)(p + 16 + khalf * 8);
  return u.v;
}
__device__ __forceinline__ v8f wmma16(v16h a, v16h b, v8f c) {
  return __builtin_amdgcn_wmma_f32_16x16x32_f16(false, a, false, b, (short)0, c,
                                                false, false);
}

// Branch-free transcendental helpers: v_exp_f32 / v_rcp_f32 trans ops
// (co-execute with WMMA per ISA 15.5), no saveexec branches like libm tanhf.
__device__ __forceinline__ float fast_sigmoid(float x) {
  return __builtin_amdgcn_rcpf(1.f + __expf(-x));
}
__device__ __forceinline__ float fast_tanh(float x) {
  // 1 - 2/(e^{2x}+1); saturates correctly at +/-inf without branches.
  return 1.f - 2.f * __builtin_amdgcn_rcpf(__expf(2.f * x) + 1.f);
}

// ---------------------------------------------------------------------------
// Prep kernels (f32 -> f16 packing / transposition)
// ---------------------------------------------------------------------------

__global__ __launch_bounds__(256) void cvt_x_kernel(const float* __restrict__ X,
                                                    _Float16* __restrict__ x16,
                                                    long rows, int E, int KX) {
  long i = (long)blockIdx.x * 256 + threadIdx.x;
  long total = rows * (long)KX;
  if (i >= total) return;
  int k = (int)(i % KX);
  long n = i / KX;
  x16[i] = (k < E) ? (_Float16)X[n * E + k] : (_Float16)0.f;
}

// w: [(E+128) x 256] row-major f32  ->  dst[256][KX+128] f16 (zeros pad E..KX)
__global__ __launch_bounds__(256) void pack_gate_kernel(const float* __restrict__ w,
                                                        _Float16* __restrict__ dst,
                                                        int E, int KX) {
  int KT = KX + 128;
  long i = (long)blockIdx.x * 256 + threadIdx.x;
  if (i >= (long)256 * KT) return;
  int k = (int)(i % KT);
  int n = (int)(i / KT);
  float v;
  if (k < E) v = w[(long)k * 256 + n];
  else if (k < KX) v = 0.f;
  else v = w[(long)(E + (k - KX)) * 256 + n];
  dst[i] = (_Float16)v;
}

// w: [(E+128) x 128] -> dst[128][KX]  (x part of candidate weights, padded)
__global__ __launch_bounds__(256) void pack_candx_kernel(const float* __restrict__ w,
                                                         _Float16* __restrict__ dst,
                                                         int E, int KX) {
  long i = (long)blockIdx.x * 256 + threadIdx.x;
  if (i >= (long)128 * KX) return;
  int k = (int)(i % KX);
  int n = (int)(i / KX);
  dst[i] = (k < E) ? (_Float16)w[(long)k * 128 + n] : (_Float16)0.f;
}

// w: [(E+128) x 128] -> dst[128][128]  (h part of candidate weights)
__global__ __launch_bounds__(256) void pack_candh_kernel(const float* __restrict__ w,
                                                         _Float16* __restrict__ dst,
                                                         int E) {
  int i = blockIdx.x * 256 + threadIdx.x;
  if (i >= 128 * 128) return;
  int k = i % 128;
  int n = i / 128;
  dst[i] = (_Float16)w[(long)(E + k) * 128 + n];
}

// wa: [256 x 128] -> dst[128][256]
__global__ __launch_bounds__(256) void pack_waT_kernel(const float* __restrict__ wa,
                                                       _Float16* __restrict__ dst) {
  int i = blockIdx.x * 256 + threadIdx.x;
  if (i >= 128 * 256) return;
  int k = i % 256;
  int a = i / 256;
  dst[i] = (_Float16)wa[(long)k * 128 + a];
}

// ---------------------------------------------------------------------------
// Batched bidirectional GRU step kernel. 16 sequences per block, 8 waves.
// Per step:  phase1: G[16,256] = [x_t | h] @ Wg,  Cx[16,128] = x_t @ Wc_x
//            phase2: c = tanh(Cx + (r*h) @ Wc_h); h = z*h + (1-z)*c
// Weights live in LDS (f16, transposed). grid = (nseq/16, 2 dirs).
// ---------------------------------------------------------------------------

template <int KX, int T>
__global__ __launch_bounds__(256) void gru_kernel(
    const _Float16* __restrict__ x16,                                   // [N][T][KX]
    const _Float16* __restrict__ wgT_f, const _Float16* __restrict__ wgT_b,  // [256][KX+128]
    const _Float16* __restrict__ wcxT_f, const _Float16* __restrict__ wcxT_b, // [128][KX]
    const _Float16* __restrict__ wchT_f, const _Float16* __restrict__ wchT_b, // [128][128]
    const float* __restrict__ bg_f, const float* __restrict__ bg_b,     // [256]
    const float* __restrict__ bc_f, const float* __restrict__ bc_b,     // [128]
    const int* __restrict__ lens,                                       // [N]
    _Float16* __restrict__ out)                                         // [N][T][256]
{
  constexpr int H = 128, NG = 256, KT = KX + H;
  __shared__ __align__(16) _Float16 sWg[NG * KT];
  __shared__ __align__(16) _Float16 sWcx[H * KX];
  __shared__ __align__(16) _Float16 sWch[H * H];
  __shared__ __align__(16) _Float16 sh[16 * H];
  __shared__ __align__(16) _Float16 srh[16 * H];
  __shared__ float sbg[NG];
  __shared__ float sbc[H];
  __shared__ int slen[16];

  const int tid = threadIdx.x;
  const int lane = tid & 31, wave = tid >> 5;
  const int mrow = lane & 15, kgrp = lane >> 4;
  const int dir = blockIdx.y;
  const int seq0 = blockIdx.x * 16;

  const _Float16* wgT = dir ? wgT_b : wgT_f;
  const _Float16* wcxT = dir ? wcxT_b : wcxT_f;
  const _Float16* wchT = dir ? wchT_b : wchT_f;
  const float* bg = dir ? bg_b : bg_f;
  const float* bc = dir ? bc_b : bc_f;

  for (int i = tid; i < NG * KT / 8; i += 256)
    ((float4*)sWg)[i] = ((const float4*)wgT)[i];
  for (int i = tid; i < H * KX / 8; i += 256)
    ((float4*)sWcx)[i] = ((const float4*)wcxT)[i];
  for (int i = tid; i < H * H / 8; i += 256)
    ((float4*)sWch)[i] = ((const float4*)wchT)[i];
  if (tid < NG) sbg[tid] = bg[tid];
  if (tid < H) sbc[tid] = bc[tid];
  if (tid < 16) slen[tid] = lens[seq0 + tid];
  for (int i = tid; i < 16 * H / 8; i += 256)
    ((float4*)sh)[i] = make_float4(0.f, 0.f, 0.f, 0.f);
  __syncthreads();

  const int nG0 = wave * 16;        // r-gate tile base; also candidate tile base
  const int nG1 = 128 + wave * 16;  // z-gate tile base

  for (int s = 0; s < T; ++s) {
    v8f aG0 = {}, aG1 = {}, aC = {};
    const int lenA = slen[mrow];
    int tA = dir ? (lenA - 1 - s) : s;
    if (tA < 0) tA = 0;  // invalid steps are masked later; keep loads in-bounds
    const _Float16* xrow = x16 + ((long)(seq0 + mrow) * T + tA) * KX;
    if (s + 1 < T) {  // prefetch next step's activations
      int tN = dir ? (lenA - 2 - s) : (s + 1);
      if (tN < 0) tN = 0;
      __builtin_prefetch(x16 + ((long)(seq0 + mrow) * T + tN) * KX, 0, 3);
    }
    // ---- phase 1: gates over [x|h], candidate x-part --------------------
    for (int kc = 0; kc < KT / 32; ++kc) {
      const int k0 = kc * 32;
      v16h a = (k0 < KX) ? fragA(xrow + k0, kgrp)
                         : fragA(sh + mrow * H + (k0 - KX), kgrp);
      aG0 = wmma16(a, fragB(sWg + (nG0 + mrow) * KT + k0, kgrp), aG0);
      aG1 = wmma16(a, fragB(sWg + (nG1 + mrow) * KT + k0, kgrp), aG1);
      if (k0 < KX)
        aC = wmma16(a, fragB(sWcx + (nG0 + mrow) * KX + k0, kgrp), aC);
    }
    // ---- gate nonlinearity: r -> r*h staged in LDS, z stays in regs -----
    float zreg[8];
    {
      const int colr = nG0 + mrow;  // < 128
      const int colz = nG1 + mrow;  // >= 128
      const float br = sbg[colr], bz = sbg[colz];
#pragma unroll
      for (int r = 0; r < 8; ++r) {
        const int m = r + 8 * kgrp;
        float rv = fast_sigmoid(aG0[r] + br);
        srh[m * H + colr] = (_Float16)(rv * (float)sh[m * H + colr]);
        zreg[r] = fast_sigmoid(aG1[r] + bz);
      }
    }
    __syncthreads();
    // ---- phase 2: candidate h-part, state update ------------------------
    for (int kc = 0; kc < H / 32; ++kc) {
      v16h a = fragA(srh + mrow * H + kc * 32, kgrp);
      v16h b = fragB(sWch + (nG0 + mrow) * H + kc * 32, kgrp);
      aC = wmma16(a, b, aC);
    }
    {
      const int col = nG0 + mrow;
      const float bcv = sbc[col];
#pragma unroll
      for (int r = 0; r < 8; ++r) {
        const int m = r + 8 * kgrp;
        float c = fast_tanh(aC[r] + bcv);
        float hold = (float)sh[m * H + col];
        float hnew = zreg[r] * hold + (1.f - zreg[r]) * c;
        const int len = slen[m];
        const bool valid = (s < len);
        float hv = valid ? hnew : hold;
        sh[m * H + col] = (_Float16)hv;
        if (valid) {
          const int t = dir ? (len - 1 - s) : s;
          out[((long)(seq0 + m) * T + t) * (2 * H) + dir * H + col] = (_Float16)hv;
        }
      }
    }
    __syncthreads();
  }
}

// ---------------------------------------------------------------------------
// Attention projection + score:  score[row] = tanh(out[row]@Wa + ba) . u
// 16 rows per block; WMMA GEMM [16,256]x[256,128] with fused epilogue.
// ---------------------------------------------------------------------------

__global__ __launch_bounds__(256) void att_proj_kernel(
    const _Float16* __restrict__ out16,  // [R][256]
    const _Float16* __restrict__ waT,    // [128][256]
    const float* __restrict__ ba,        // [128]
    const float* __restrict__ u,         // [128]
    float* __restrict__ scores)          // [R]
{
  __shared__ __align__(16) _Float16 sWa[128 * 256];
  __shared__ float sba[128], su[128];
  __shared__ float sred[8][16];
  const int tid = threadIdx.x;
  const int lane = tid & 31, wave = tid >> 5;
  const int mrow = lane & 15, kgrp = lane >> 4;
  for (int i = tid; i < 128 * 256 / 8; i += 256)
    ((float4*)sWa)[i] = ((const float4*)waT)[i];
  if (tid < 128) { sba[tid] = ba[tid]; su[tid] = u[tid]; }
  __syncthreads();

  const long row0 = (long)blockIdx.x * 16;
  const _Float16* arow = out16 + (row0 + mrow) * 256;
  v8f acc = {};
  for (int kc = 0; kc < 8; ++kc) {
    v16h a = fragA(arow + kc * 32, kgrp);
    v16h b = fragB(sWa + (wave * 16 + mrow) * 256 + kc * 32, kgrp);
    acc = wmma16(a, b, acc);
  }
  const int col = wave * 16 + mrow;
  const float bav = sba[col], uv = su[col];
  float part[8];
#pragma unroll
  for (int r = 0; r < 8; ++r) part[r] = fast_tanh(acc[r] + bav) * uv;
#pragma unroll
  for (int r = 0; r < 8; ++r) {  // reduce over the 16 lanes of each half
    float v = part[r];
    for (int off = 1; off < 16; off <<= 1) v += __shfl_xor(v, off, 32);
    part[r] = v;
  }
  if ((lane & 15) == 0) {
#pragma unroll
    for (int r = 0; r < 8; ++r) sred[wave][r + 8 * kgrp] = part[r];
  }
  __syncthreads();
  if (tid < 16) {
    float s = 0.f;
    for (int w = 0; w < 8; ++w) s += sred[w][tid];
    scores[row0 + tid] = s;  // DISTILL == 1.0
  }
}

// ---------------------------------------------------------------------------
// Per-sequence softmax over T and attention-weighted pooling.
// ---------------------------------------------------------------------------

template <int T>
__global__ __launch_bounds__(256) void att_pool_kernel(
    const float* __restrict__ scores,   // [N*T]
    const _Float16* __restrict__ out16, // [N][T][256]
    float* __restrict__ att_out,        // [N*T]
    _Float16* __restrict__ vec16,       // [N][256] or nullptr
    float* __restrict__ vec32)          // [N][256] or nullptr
{
  __shared__ float satt[T];
  const int n = blockIdx.x, tid = threadIdx.x;
  if (tid < T) satt[tid] = scores[(long)n * T + tid];
  __syncthreads();
  if (tid == 0) {
    float mx = -1e30f;
    for (int t = 0; t < T; ++t) mx = fmaxf(mx, satt[t]);
    float sm = 0.f;
    for (int t = 0; t < T; ++t) { float e = __expf(satt[t] - mx); satt[t] = e; sm += e; }
    float inv = 1.f / sm;
    for (int t = 0; t < T; ++t) satt[t] *= inv;
  }
  __syncthreads();
  if (tid < T) att_out[(long)n * T + tid] = satt[tid];
  float acc = 0.f;
  const _Float16* base = out16 + (long)n * T * 256 + tid;
  for (int t = 0; t < T; ++t) acc += satt[t] * (float)base[t * 256];
  if (vec16) vec16[(long)n * 256 + tid] = (_Float16)acc;
  if (vec32) vec32[(long)n * 256 + tid] = acc;
}

// ---------------------------------------------------------------------------
// Head: logits, log-softmax loss (+ regularizer), predict, accuracy.
// ---------------------------------------------------------------------------

__global__ __launch_bounds__(256) void final_kernel(
    const float* __restrict__ docvec,  // [64][256]
    const float* __restrict__ wp,      // [256][10]
    const float* __restrict__ bp,      // [10]
    const int* __restrict__ y,         // [64]
    const float* __restrict__ wa_l, const float* __restrict__ ba_l,
    const float* __restrict__ wa_h,
    const float* __restrict__ uw, const float* __restrict__ us,
    float* __restrict__ dout)
{
  __shared__ float slog[64][10];
  __shared__ float sred[256];
  __shared__ float sloss[64];
  __shared__ float sacc[64];
  const int tid = threadIdx.x;
  for (int e = tid; e < 640; e += 256) {
    const int rr = e / 10, c = e % 10;
    float a = bp[c];
    for (int k = 0; k < 256; ++k) a += docvec[rr * 256 + k] * wp[k * 10 + c];
    slog[rr][c] = a;
  }
  float rs = 0.f;
  for (int i = tid; i < 256 * 128; i += 256) {
    float v = wa_l[i]; rs += v * v;
    float w = wa_h[i]; rs += 2.f * w * w;  // reference counts wa_h twice
  }
  for (int i = tid; i < 128; i += 256) {
    float v = ba_l[i]; rs += v * v;
    float a = uw[i]; rs += a * a;
    float b = us[i]; rs += b * b;
  }
  sred[tid] = rs;
  __syncthreads();
  for (int st = 128; st > 0; st >>= 1) {
    if (tid < st) sred[tid] += sred[tid + st];
    __syncthreads();
  }
  if (tid < 64) {
    float mx = -1e30f; int am = 0;
    for (int c = 0; c < 10; ++c)
      if (slog[tid][c] > mx) { mx = slog[tid][c]; am = c; }
    float sm = 0.f;
    for (int c = 0; c < 10; ++c) sm += __expf(slog[tid][c] - mx);
    const float lse = mx + __logf(sm);
    const int yv = y[tid];
    sloss[tid] = -(slog[tid][yv] - lse);
    sacc[tid] = (am == yv) ? 1.f : 0.f;
    dout[1 + tid] = (float)am;
  }
  __syncthreads();
  if (tid == 0) {
    float L = 0.f, A = 0.f;
    for (int r = 0; r < 64; ++r) { L += sloss[r]; A += sacc[r]; }
    dout[0] = L / 64.f + 0.01f * sred[0];
    dout[65] = A / 64.f;
  }
}

// ---------------------------------------------------------------------------
// Host orchestration
// ---------------------------------------------------------------------------

extern "C" void kernel_launch(void* const* d_in, const int* in_sizes, int n_in,
                              void* d_out, int out_size, void* d_ws, size_t ws_size,
                              hipStream_t stream) {
  (void)in_sizes; (void)n_in; (void)out_size; (void)ws_size;
  const float* X = (const float*)d_in[0];
  const int* y = (const int*)d_in[1];
  const int* slen = (const int*)d_in[2];
  const int* dlen = (const int*)d_in[3];
  const float* wgf_l = (const float*)d_in[6];  const float* bgf_l = (const float*)d_in[7];
  const float* wcf_l = (const float*)d_in[8];  const float* bcf_l = (const float*)d_in[9];
  const float* wgb_l = (const float*)d_in[10]; const float* bgb_l = (const float*)d_in[11];
  const float* wcb_l = (const float*)d_in[12]; const float* bcb_l = (const float*)d_in[13];
  const float* wa_l = (const float*)d_in[14];  const float* ba_l = (const float*)d_in[15];
  const float* wgf_h = (const float*)d_in[16]; const float* bgf_h = (const float*)d_in[17];
  const float* wcf_h = (const float*)d_in[18]; const float* bcf_h = (const float*)d_in[19];
  const float* wgb_h = (const float*)d_in[20]; const float* bgb_h = (const float*)d_in[21];
  const float* wcb_h = (const float*)d_in[22]; const float* bcb_h = (const float*)d_in[23];
  const float* wa_h = (const float*)d_in[24];  const float* ba_h = (const float*)d_in[25];
  const float* uw = (const float*)d_in[26];    const float* us = (const float*)d_in[27];
  const float* wp = (const float*)d_in[28];    const float* bp = (const float*)d_in[29];
  float* dout = (float*)d_out;

  // ---- workspace layout (512B aligned slabs) -----------------------------
  char* ws = (char*)d_ws;
  size_t o = 0;
  auto alloc = [&](size_t b) { size_t r = o; o = (o + b + 511) & ~(size_t)511; return r; };
  const long NL = 2560;           // word-level sequences
  const size_t off_x16     = alloc((size_t)NL * 50 * 224 * 2);
  const size_t off_wgT_lf  = alloc(256 * 352 * 2);
  const size_t off_wgT_lb  = alloc(256 * 352 * 2);
  const size_t off_wcxT_lf = alloc(128 * 224 * 2);
  const size_t off_wcxT_lb = alloc(128 * 224 * 2);
  const size_t off_wchT_lf = alloc(128 * 128 * 2);
  const size_t off_wchT_lb = alloc(128 * 128 * 2);
  const size_t off_wgT_hf  = alloc(256 * 384 * 2);
  const size_t off_wgT_hb  = alloc(256 * 384 * 2);
  const size_t off_wcxT_hf = alloc(128 * 256 * 2);
  const size_t off_wcxT_hb = alloc(128 * 256 * 2);
  const size_t off_wchT_hf = alloc(128 * 128 * 2);
  const size_t off_wchT_hb = alloc(128 * 128 * 2);
  const size_t off_waT_l   = alloc(128 * 256 * 2);
  const size_t off_waT_h   = alloc(128 * 256 * 2);
  const size_t off_outlo   = alloc((size_t)NL * 50 * 256 * 2);
  const size_t off_outhi   = alloc((size_t)64 * 40 * 256 * 2);
  const size_t off_sent16  = alloc((size_t)64 * 40 * 256 * 2);
  const size_t off_sc_lo   = alloc((size_t)NL * 50 * 4);
  const size_t off_sc_hi   = alloc((size_t)64 * 40 * 4);
  const size_t off_docvec  = alloc((size_t)64 * 256 * 4);

  _Float16* x16 = (_Float16*)(ws + off_x16);
  _Float16* wgT_lf = (_Float16*)(ws + off_wgT_lf);
  _Float16* wgT_lb = (_Float16*)(ws + off_wgT_lb);
  _Float16* wcxT_lf = (_Float16*)(ws + off_wcxT_lf);
  _Float16* wcxT_lb = (_Float16*)(ws + off_wcxT_lb);
  _Float16* wchT_lf = (_Float16*)(ws + off_wchT_lf);
  _Float16* wchT_lb = (_Float16*)(ws + off_wchT_lb);
  _Float16* wgT_hf = (_Float16*)(ws + off_wgT_hf);
  _Float16* wgT_hb = (_Float16*)(ws + off_wgT_hb);
  _Float16* wcxT_hf = (_Float16*)(ws + off_wcxT_hf);
  _Float16* wcxT_hb = (_Float16*)(ws + off_wcxT_hb);
  _Float16* wchT_hf = (_Float16*)(ws + off_wchT_hf);
  _Float16* wchT_hb = (_Float16*)(ws + off_wchT_hb);
  _Float16* waT_l = (_Float16*)(ws + off_waT_l);
  _Float16* waT_h = (_Float16*)(ws + off_waT_h);
  _Float16* out_lo = (_Float16*)(ws + off_outlo);
  _Float16* out_hi = (_Float16*)(ws + off_outhi);
  _Float16* sent16 = (_Float16*)(ws + off_sent16);
  float* sc_lo = (float*)(ws + off_sc_lo);
  float* sc_hi = (float*)(ws + off_sc_hi);
  float* docvec = (float*)(ws + off_docvec);

  // ---- prep --------------------------------------------------------------
  hipMemsetAsync(out_lo, 0, (size_t)NL * 50 * 256 * 2, stream);
  hipMemsetAsync(out_hi, 0, (size_t)64 * 40 * 256 * 2, stream);

  {
    long total = (long)NL * 50 * 224;
    cvt_x_kernel<<<(unsigned)((total + 255) / 256), 256, 0, stream>>>(
        X, x16, NL * 50, 200, 224);
  }
  pack_gate_kernel<<<(256 * 352 + 255) / 256, 256, 0, stream>>>(wgf_l, wgT_lf, 200, 224);
  pack_gate_kernel<<<(256 * 352 + 255) / 256, 256, 0, stream>>>(wgb_l, wgT_lb, 200, 224);
  pack_candx_kernel<<<(128 * 224 + 255) / 256, 256, 0, stream>>>(wcf_l, wcxT_lf, 200, 224);
  pack_candx_kernel<<<(128 * 224 + 255) / 256, 256, 0, stream>>>(wcb_l, wcxT_lb, 200, 224);
  pack_candh_kernel<<<(128 * 128 + 255) / 256, 256, 0, stream>>>(wcf_l, wchT_lf, 200);
  pack_candh_kernel<<<(128 * 128 + 255) / 256, 256, 0, stream>>>(wcb_l, wchT_lb, 200);
  pack_gate_kernel<<<(256 * 384 + 255) / 256, 256, 0, stream>>>(wgf_h, wgT_hf, 256, 256);
  pack_gate_kernel<<<(256 * 384 + 255) / 256, 256, 0, stream>>>(wgb_h, wgT_hb, 256, 256);
  pack_candx_kernel<<<(128 * 256 + 255) / 256, 256, 0, stream>>>(wcf_h, wcxT_hf, 256, 256);
  pack_candx_kernel<<<(128 * 256 + 255) / 256, 256, 0, stream>>>(wcb_h, wcxT_hb, 256, 256);
  pack_candh_kernel<<<(128 * 128 + 255) / 256, 256, 0, stream>>>(wcf_h, wchT_hf, 256);
  pack_candh_kernel<<<(128 * 128 + 255) / 256, 256, 0, stream>>>(wcb_h, wchT_hb, 256);
  pack_waT_kernel<<<(128 * 256 + 255) / 256, 256, 0, stream>>>(wa_l, waT_l);
  pack_waT_kernel<<<(128 * 256 + 255) / 256, 256, 0, stream>>>(wa_h, waT_h);

  // ---- word level --------------------------------------------------------
  gru_kernel<224, 50><<<dim3(2560 / 16, 2), 256, 0, stream>>>(
      x16, wgT_lf, wgT_lb, wcxT_lf, wcxT_lb, wchT_lf, wchT_lb,
      bgf_l, bgb_l, bcf_l, bcb_l, slen, out_lo);
  att_proj_kernel<<<(2560 * 50) / 16, 256, 0, stream>>>(out_lo, waT_l, ba_l, uw, sc_lo);
  att_pool_kernel<50><<<2560, 256, 0, stream>>>(
      sc_lo, out_lo, dout + 66, sent16, (float*)nullptr);

  // ---- sentence level ----------------------------------------------------
  gru_kernel<256, 40><<<dim3(64 / 16, 2), 256, 0, stream>>>(
      sent16, wgT_hf, wgT_hb, wcxT_hf, wcxT_hb, wchT_hf, wchT_hb,
      bgf_h, bgb_h, bcf_h, bcb_h, dlen, out_hi);
  att_proj_kernel<<<(64 * 40) / 16, 256, 0, stream>>>(out_hi, waT_h, ba_h, us, sc_hi);
  att_pool_kernel<40><<<64, 256, 0, stream>>>(
      sc_hi, out_hi, dout + 66 + 2560 * 50, (_Float16*)nullptr, docvec);

  // ---- head --------------------------------------------------------------
  final_kernel<<<1, 256, 0, stream>>>(docvec, wp, bp, y, wa_l, ba_l, wa_h, uw, us, dout);
}
